// MixtureOfExperts_63771674411213
// MI455X (gfx1250) — matrix operand order repaired and checked by reading, not compiled, not run on hardware
//
#include <hip/hip_runtime.h>
#include <hip/hip_bf16.h>
#include <math.h>

// ---------------- problem constants (match reference) ----------------
#define BATCH     65536
#define FEAT      512
#define NEXP      8
#define NCLS      125      // classes per expert
#define NPAD      128      // NCLS padded to 8 WMMA n-tiles
#define MT        64       // rows per expert-GEMM block (4 x 16-row WMMA tiles)
#define KCHUNK    32
#define NKCH      (FEAT / KCHUNK)          // 16 K-chunks
#define CHUNK_ELEMS (NPAD * KCHUNK)        // 4096 bf16 = 8 KB per staged chunk

// ---------------- vector types ----------------
typedef __bf16          v16bf  __attribute__((ext_vector_type(16)));
typedef unsigned short  v16u16 __attribute__((ext_vector_type(16)));
typedef float           v8f    __attribute__((ext_vector_type(8)));
typedef unsigned int    v4u    __attribute__((ext_vector_type(4)));
typedef int             v8i_   __attribute__((ext_vector_type(8)));
typedef int             v4i_   __attribute__((ext_vector_type(4)));

__device__ __forceinline__ unsigned short f32_to_bf16_bits(float f) {
    unsigned int u = __float_as_uint(f);
    unsigned int r = u + 0x7FFFu + ((u >> 16) & 1u);
    return (unsigned short)(r >> 16);
}

// ============================================================
// 0) zero the atomic counters (counts / offsets / cursors)
// ============================================================
__global__ void moe_init_kernel(int* wsi) {
    if (threadIdx.x < 24) wsi[threadIdx.x] = 0;
}

// ============================================================
// 0b) one-time prep: We (f32, [8][512][125]) -> pre-tiled bf16
//     Wtile[e][c][col][k], each (e,c) chunk = contiguous 8 KB,
//     exactly the LDS image the WMMA B-fragments want.
// ============================================================
__global__ __launch_bounds__(256) void moe_prep_kernel(
    const float* __restrict__ We, unsigned short* __restrict__ Wtile)
{
    int id  = blockIdx.x * 256 + threadIdx.x;   // 8*16*128*32 = 524288 ids
    int k   = id & 31;
    int col = (id >> 5) & 127;
    int c   = (id >> 12) & 15;
    int e   = id >> 16;
    int cc  = col < NCLS ? col : NCLS - 1;      // clamp pad cols (never stored)
    float w = We[((size_t)e * FEAT + (c * KCHUNK + k)) * NCLS + cc];
    Wtile[id] = f32_to_bf16_bits(w);
}

// ============================================================
// 1) coarse gating: one wave per row.  coarse = X@Wc + bc,
//    argmax -> expert_id, atomic per-expert histogram.
// ============================================================
__global__ __launch_bounds__(256) void moe_coarse_kernel(
    const float* __restrict__ feat, const float* __restrict__ Wc,
    const float* __restrict__ bc, float* __restrict__ coarse,
    float* __restrict__ expert_out, int* __restrict__ eid,
    int* __restrict__ counts)
{
    __shared__ float WcT[NEXP * FEAT];   // transposed: WcT[e*512 + k], 16 KB
    const int tid = threadIdx.x;
    for (int i = tid; i < NEXP * FEAT; i += 256) {
        int e = i >> 9, k = i & (FEAT - 1);
        WcT[i] = Wc[k * NEXP + e];
    }
    __syncthreads();

    const int wave = tid >> 5, lane = tid & 31;
    const int row  = blockIdx.x * 8 + wave;
    const float* x = feat + (size_t)row * FEAT;

    float acc[NEXP];
    #pragma unroll
    for (int e = 0; e < NEXP; ++e) acc[e] = 0.f;

    #pragma unroll
    for (int j = 0; j < FEAT / 32; ++j) {
        int k   = lane + 32 * j;
        float v = x[k];
        #pragma unroll
        for (int e = 0; e < NEXP; ++e) acc[e] = fmaf(v, WcT[e * FEAT + k], acc[e]);
    }
    #pragma unroll
    for (int off = 16; off >= 1; off >>= 1) {
        #pragma unroll
        for (int e = 0; e < NEXP; ++e) acc[e] += __shfl_xor(acc[e], off, 32);
    }
    if (lane == 0) {
        float best = -1e30f; int bi = 0;
        #pragma unroll
        for (int e = 0; e < NEXP; ++e) {
            float v = acc[e] + bc[e];
            coarse[(size_t)row * NEXP + e] = v;
            if (v > best) { best = v; bi = e; }   // strict '>' == first-max (JAX tie rule)
        }
        expert_out[row] = (float)bi;
        eid[row] = bi;
        atomicAdd(&counts[bi], 1);
    }
}

// ============================================================
// 2) exclusive scan of the 8 expert counts
// ============================================================
__global__ void moe_scan_kernel(int* wsi) {
    if (threadIdx.x == 0) {
        int acc = 0;
        for (int e = 0; e < NEXP; ++e) {
            wsi[8 + e]  = acc;   // offsets
            wsi[16 + e] = acc;   // cursors
            acc += wsi[e];
        }
    }
}

// ============================================================
// 3) scatter rows into per-expert contiguous index groups
// ============================================================
__global__ __launch_bounds__(256) void moe_scatter_kernel(
    const int* __restrict__ eid, int* __restrict__ cursors,
    int* __restrict__ perm)
{
    int i = blockIdx.x * 256 + threadIdx.x;
    int e = eid[i];
    int pos = atomicAdd(&cursors[e], 1);
    perm[pos] = i;
}

// ============================================================
// 4) routed expert head via bf16 WMMA (f32 accumulate).
//    B chunks staged global->LDS with the Tensor Data Mover,
//    fused bias + softmax + argmax epilogue.
// ============================================================
__global__ __launch_bounds__(256) void moe_expert_kernel(
    const float* __restrict__ feat, const unsigned short* __restrict__ Wtile,
    const float* __restrict__ be,   const int* __restrict__ counts,
    const int* __restrict__ offsets, const int* __restrict__ perm,
    float* __restrict__ local_preds, float* __restrict__ global_preds)
{
    __shared__ alignas(32) unsigned short Bsh[NPAD][KCHUNK]; // bf16 chunk [col][k], 8 KB
    __shared__ float Lsh[MT][NPAD];                          // logits, 32 KB

    // ---- map block -> (expert, tile) over variable-size groups ----
    const int bid = blockIdx.x;
    int e = -1, tile = 0, accb = 0;
    #pragma unroll
    for (int i = 0; i < NEXP; ++i) {
        int nb = (counts[i] + MT - 1) / MT;
        if (e < 0 && bid < accb + nb) { e = i; tile = bid - accb; }
        accb += nb;
    }
    if (e < 0) return;                       // uniform per-block: safe vs barriers

    const int cnt   = counts[e];
    const int gbase = offsets[e] + tile * MT;
    const int tid   = threadIdx.x;
    const int wave  = tid >> 5, lane = tid & 31;
    const int m_sub = wave >> 1;             // which 16-row tile
    const int n_half = wave & 1;             // cols 0..63 or 64..127

    // A-fragment row for this lane (16-bit A 16x32 layout: lanes 0-15 / 16-31 split K)
    const int rloc  = m_sub * 16 + (lane & 15);
    const bool valid = (tile * MT + rloc) < cnt;
    const int gidx  = valid ? (gbase + rloc) : (offsets[e] + cnt - 1);
    const int srcrow = perm[gidx];
    const float* Arow = feat + (size_t)srcrow * FEAT;
    const int kA = (lane >> 4) << 4;         // 0 or 16
    const int kB = (lane >> 4) << 4;         // 0 or 16 (contiguous k = 16h + s, same order as A)

    v8f acc[4];
    #pragma unroll
    for (int t = 0; t < 4; ++t) acc[t] = v8f{0.f,0.f,0.f,0.f,0.f,0.f,0.f,0.f};

    for (int c = 0; c < NKCH; ++c) {
        const int kc = c * KCHUNK;
        const unsigned short* chunk = Wtile + (size_t)(e * NKCH + c) * CHUNK_ELEMS;
        __syncthreads();   // previous chunk fully consumed before overwrite

#if __has_builtin(__builtin_amdgcn_tensor_load_to_lds)
        // ---- TDM: one 8 KB contiguous bf16 tile, global -> LDS ----
        if (wave == 0) {
            unsigned long long ga = (unsigned long long)(uintptr_t)chunk;
            unsigned ldsa = (unsigned)(uintptr_t)(&Bsh[0][0]);
            v4u  g0 = {0u, 0u, 0u, 0u};
            v8i_ g1 = {0, 0, 0, 0, 0, 0, 0, 0};
            v4i_ gz = {0, 0, 0, 0};
            g0[0] = 1u;                                   // count=1, user descriptor
            g0[1] = ldsa;                                 // LDS dest (bytes)
            g0[2] = (unsigned)(ga & 0xFFFFFFFFu);         // global addr lo
            g0[3] = (unsigned)((ga >> 32) & 0x01FFFFFFu)  // global addr hi (57-bit)
                    | (2u << 30);                         // type = 2 (image)
            g1[0] = (int)(1u << 16);                      // data_size = 1 (2 bytes)
            g1[1] = (int)((CHUNK_ELEMS & 0xFFFFu) << 16); // tensor_dim0 lo16
            g1[2] = (int)(((CHUNK_ELEMS >> 16) & 0xFFFFu) // tensor_dim0 hi16
                    | (1u << 16));                        // tensor_dim1 = 1
            g1[3] = (int)((unsigned)CHUNK_ELEMS << 16);   // tile_dim0 = 4096
            g1[4] = 1;                                    // tile_dim1 = 1
            g1[5] = (int)CHUNK_ELEMS;                     // tensor_dim0_stride
            g1[6] = 0;
            g1[7] = 0;
#if __clang_major__ >= 23
            v8i_ gz8 = {0, 0, 0, 0, 0, 0, 0, 0};
            __builtin_amdgcn_tensor_load_to_lds(g0, g1, gz, gz, gz8, 0);
#else
            __builtin_amdgcn_tensor_load_to_lds(g0, g1, gz, gz, 0);
#endif
#if __has_builtin(__builtin_amdgcn_s_wait_tensorcnt)
            __builtin_amdgcn_s_wait_tensorcnt(0);
#else
            asm volatile("s_wait_tensorcnt 0x0" ::: "memory");
#endif
        }
#else
        // ---- fallback: cooperative 16B copies of the same pre-tiled blob ----
        {
            uint4* dst = (uint4*)&Bsh[0][0];
            const uint4* src = (const uint4*)chunk;
            dst[tid]       = src[tid];
            dst[tid + 256] = src[tid + 256];
        }
#endif
        __syncthreads();

        // ---- A fragment: f32 global gather -> bf16 (hw cvt) ----
        const float* ap = Arow + kc + kA;
        __builtin_prefetch(ap + 32, 0, 1);   // next K-chunk -> global_prefetch_b8
        v16bf a;
        #pragma unroll
        for (int j = 0; j < 8; ++j) a[j]     = (__bf16)ap[j];
        #pragma unroll
        for (int j = 0; j < 8; ++j) a[8 + j] = (__bf16)ap[8 + j];

        // ---- 4 n-tiles: B frag from LDS (contiguous 32B), WMMA accumulate ----
        #pragma unroll
        for (int t = 0; t < 4; ++t) {
            int col = n_half * 64 + t * 16 + (lane & 15);
            v16u16 bu = *(const v16u16*)&Bsh[col][kB];
            v16bf b = __builtin_bit_cast(v16bf, bu);
            acc[t] = __builtin_amdgcn_wmma_f32_16x16x32_bf16(
                         false, a, false, b, (short)0, acc[t], false, false);
        }
    }

    // ---- epilogue: +bias, scatter accumulators to LDS logits ----
    __syncthreads();
    #pragma unroll
    for (int t = 0; t < 4; ++t) {
        int col   = n_half * 64 + t * 16 + (lane & 15);
        int rbase = m_sub * 16 + ((lane >> 4) << 3);    // C/D layout: M = v (+8 hi lanes)
        float bias = (col < NCLS) ? be[e * NCLS + col] : 0.f;
        #pragma unroll
        for (int v = 0; v < 8; ++v)
            Lsh[rbase + v][col] = acc[t][v] + bias;
    }
    __syncthreads();

    // ---- per-row softmax + argmax, write outputs ----
    if (tid < MT) {
        int r = tid;
        if (tile * MT + r < cnt) {
            int row = perm[gbase + r];
            float mx = -1e30f; int am = 0;
            for (int j = 0; j < NCLS; ++j) {
                float x = Lsh[r][j];
                if (x > mx) { mx = x; am = j; }
            }
            float s = 0.f;
            for (int j = 0; j < NCLS; ++j) s += __expf(Lsh[r][j] - mx);
            float inv = 1.f / s;
            float* out = local_preds + (size_t)row * NCLS;
            for (int j = 0; j < NCLS; ++j) out[j] = __expf(Lsh[r][j] - mx) * inv;
            global_preds[row] = (float)(e * NCLS + am);
        }
    }
}

// ============================================================
// launch
// ============================================================
extern "C" void kernel_launch(void* const* d_in, const int* in_sizes, int n_in,
                              void* d_out, int out_size, void* d_ws, size_t ws_size,
                              hipStream_t stream) {
    const float* feat = (const float*)d_in[0];   // [B,512]
    const float* Wc   = (const float*)d_in[1];   // [512,8]
    const float* bc   = (const float*)d_in[2];   // [8]
    const float* We   = (const float*)d_in[3];   // [8,512,125]
    const float* be   = (const float*)d_in[4];   // [8,125]

    float* out          = (float*)d_out;
    float* coarse       = out;                                     // B*8
    float* expert_out   = out + (size_t)BATCH * NEXP;              // B
    float* local_preds  = out + (size_t)BATCH * (NEXP + 1);        // B*125
    float* global_preds = out + (size_t)BATCH * (NEXP + 1 + NCLS); // B

    // ws layout: [0..7] counts, [8..15] offsets, [16..23] cursors,
    //            [32..32+B) expert_id, [32+B..32+2B) perm,
    //            then 1 MB pre-tiled bf16 We   (~1.5 MB total)
    int* wsi     = (int*)d_ws;
    int* counts  = wsi;
    int* offsets = wsi + 8;
    int* cursors = wsi + 16;
    int* eid     = wsi + 32;
    int* perm    = wsi + 32 + BATCH;
    unsigned short* Wtile = (unsigned short*)(wsi + 32 + 2 * BATCH);

    moe_init_kernel   <<<1, 32, 0, stream>>>(wsi);
    moe_prep_kernel   <<<(NEXP * NKCH * CHUNK_ELEMS) / 256, 256, 0, stream>>>(We, Wtile);
    moe_coarse_kernel <<<BATCH / 8, 256, 0, stream>>>(feat, Wc, bc, coarse,
                                                      expert_out, eid, counts);
    moe_scan_kernel   <<<1, 32, 0, stream>>>(wsi);
    moe_scatter_kernel<<<BATCH / 256, 256, 0, stream>>>(eid, cursors, perm);
    moe_expert_kernel <<<BATCH / MT + NEXP, 256, 0, stream>>>(feat, Wtile, be, counts,
                                                              offsets, perm,
                                                              local_preds, global_preds);
}
